// SpatialMultiHeadSelfAttention_14199161880712
// MI455X (gfx1250) — matrix operand order repaired and checked by reading, not compiled, not run on hardware
//
#include <hip/hip_runtime.h>
#include <hip/hip_bf16.h>

typedef __attribute__((ext_vector_type(16))) __bf16 v16bf;
typedef __attribute__((ext_vector_type(2)))  __bf16 v2bf;
typedef __attribute__((ext_vector_type(8)))  float  v8f;

#define NB   2
#define DIM  256
#define NH   8
#define HD   32
#define NPIX 4096
#define LOG2E 1.4426950408889634f

// Native bf16 conversion: let the backend pick v_cvt_pk_bf16_f32.
__device__ __forceinline__ unsigned pack2(float a, float b) {
    union { v2bf v; unsigned u; } c;
    c.v[0] = (__bf16)a; c.v[1] = (__bf16)b;
    return c.u;
}

union BF16Frag { unsigned u[8]; v16bf v; };
union U4p      { unsigned u[4]; uint4 q; };

// ---------------------------------------------------------------------------
// Kernel 1: QKV projection GEMM.  qkv[o,(b,n)] = w_qkv[o,:] . x[b,:,n] + bias
// Double-buffered LDS, register prefetch, one barrier per K-step.
// Block tile: 64 (M) x 32 (cols), 8 waves = 4x2 of 16x16 WMMA tiles.
// ---------------------------------------------------------------------------
__global__ __launch_bounds__(256) void qkv_gemm(
    const float* __restrict__ x, const float* __restrict__ w,
    const float* __restrict__ bias,
    unsigned short* __restrict__ qb, unsigned short* __restrict__ kb,
    unsigned short* __restrict__ vb)
{
    __shared__ unsigned short lds_a[2][64 * 40];   // [row][k], bf16
    __shared__ unsigned short lds_b[2][32 * 40];   // [col][k], bf16 (transposed)

    const int tid  = threadIdx.x;
    const int wid  = tid >> 5, lane = tid & 31;
    const int l16  = lane & 15;
    const bool hi  = lane >= 16;
    const int b    = blockIdx.z;
    const int nbase = blockIdx.x * 32;
    const int m0    = blockIdx.y * 64;
    const int mw    = (wid >> 1) * 16;
    const int cw    = (wid & 1) * 16;

    const int arow = tid >> 2;         // 0..63
    const int akg  = (tid & 3) * 8;    // k offset 0,8,16,24
    const int bn   = tid & 31;         // col 0..31
    const int bdg  = (tid >> 5) * 4;   // d offset 0..28

    const float* asrc = w + (size_t)(m0 + arow) * DIM + akg;           // + kc
    const float* xs   = x + ((size_t)b * DIM + bdg) * NPIX + nbase + bn; // + (kc+i)*NPIX

    auto stageA = [&](int buf, float4 f0, float4 f1) {
        unsigned* dst = (unsigned*)lds_a[buf] + arow * 20 + (akg >> 1);
        dst[0] = pack2(f0.x, f0.y); dst[1] = pack2(f0.z, f0.w);
        dst[2] = pack2(f1.x, f1.y); dst[3] = pack2(f1.z, f1.w);
    };
    auto stageB = [&](int buf, float t0, float t1, float t2, float t3) {
        unsigned* dst = (unsigned*)lds_b[buf] + bn * 20 + (bdg >> 1);
        dst[0] = pack2(t0, t1); dst[1] = pack2(t2, t3);
    };

    v8f acc = {};
    float4 fa0 = *(const float4*)(asrc);
    float4 fa1 = *(const float4*)(asrc + 4);
    float bt0 = xs[0 * NPIX], bt1 = xs[1 * NPIX], bt2 = xs[2 * NPIX], bt3 = xs[3 * NPIX];
    stageA(0, fa0, fa1); stageB(0, bt0, bt1, bt2, bt3);
    __syncthreads();

    #pragma unroll 2
    for (int kci = 0; kci < 8; ++kci) {
        const int cur = kci & 1;
        if (kci < 7) {                         // prefetch next K-chunk into regs
            const int kc = (kci + 1) * 32;
            fa0 = *(const float4*)(asrc + kc);
            fa1 = *(const float4*)(asrc + kc + 4);
            bt0 = xs[(kc + 0) * NPIX]; bt1 = xs[(kc + 1) * NPIX];
            bt2 = xs[(kc + 2) * NPIX]; bt3 = xs[(kc + 3) * NPIX];
        }
        BF16Frag af, bfr;
        {   // A frag: row = mw+l16; e0..7 -> k 0..7 (+8 hi), e8..15 -> k 16..23 (+8)
            const int o4 = hi ? 4 : 0;
            const unsigned* p = (const unsigned*)lds_a[cur] + (mw + l16) * 20;
            #pragma unroll
            for (int i = 0; i < 4; i++) { af.u[i] = p[o4 + i]; af.u[4 + i] = p[8 + o4 + i]; }
        }
        {   // B frag: col = cw+l16; e0..7 -> k 0..7 (+8 hi), e8..15 -> k 16..23 (+8)
            const unsigned* p = (const unsigned*)lds_b[cur] + (cw + l16) * 20 + (hi ? 4 : 0);
            #pragma unroll
            for (int i = 0; i < 8; i++) bfr.u[i] = p[i + (i & 4)];
        }
        acc = __builtin_amdgcn_wmma_f32_16x16x32_bf16(false, af.v, false, bfr.v,
                                                      (short)0, acc, false, false);
        if (kci < 7) { stageA(cur ^ 1, fa0, fa1); stageB(cur ^ 1, bt0, bt1, bt2, bt3); }
        __syncthreads();
    }

    // Epilogue: D layout -> o = obase + r (contiguous 8-run), n = col
    const int obase = m0 + mw + (hi ? 8 : 0);
    const int n     = nbase + cw + l16;
    const int t     = obase >> 8;           // 0=q 1=k 2=v
    const int h     = (obase >> 5) & 7;
    const int e0    = obase & 31;
    unsigned short* outp = (t == 0) ? qb : ((t == 1) ? kb : vb);
    U4p pk;
    #pragma unroll
    for (int r = 0; r < 8; r += 2)
        pk.u[r >> 1] = pack2(acc[r]     + bias[obase + r],
                             acc[r + 1] + bias[obase + r + 1]);
    *(uint4*)(outp + (((size_t)(b * NH + h) * NPIX + n) * HD + e0)) = pk.q;
}

// ---------------------------------------------------------------------------
// Kernel 2: flash attention per (b,h).  Block = 128 queries (16 per wave).
// head_dim = 32 -> one WMMA K-step.  32-key chunks, double-buffered K/V tiles.
// ---------------------------------------------------------------------------
__global__ __launch_bounds__(256) void attn_kernel(
    const unsigned short* __restrict__ qbuf,
    const unsigned short* __restrict__ kbuf,
    const unsigned short* __restrict__ vbuf,
    unsigned short* __restrict__ attn_out)
{
    __shared__ unsigned short lds_k[2][32 * 40];   // [key][hd]
    __shared__ unsigned short lds_v[2][32 * 40];   // [hd][key] (transposed)
    __shared__ unsigned short pbuf[8 * 16 * 40];   // per-wave P scratch [row][key]

    const int tid = threadIdx.x, wid = tid >> 5, lane = tid & 31;
    const int l16 = lane & 15;
    const bool hi = lane >= 16;
    const int b = blockIdx.z, h = blockIdx.y;
    const int bh = b * NH + h;
    const int q0 = blockIdx.x * 128 + wid * 16;
    const float scale = 0.17677669529663687f;  // 1/sqrt(32)

    // Q A-fragment (row = query q0+l16; elements = hd dims per A layout)
    BF16Frag aq;
    {
        const uint4* qp = (const uint4*)(qbuf + ((size_t)bh * NPIX + q0 + l16) * HD);
        uint4 qa = qp[hi ? 1 : 0];   // halves 0..7  (lo) / 8..15  (hi)
        uint4 qc = qp[hi ? 3 : 2];   // halves 16..23(lo) / 24..31 (hi)
        aq.u[0] = qa.x; aq.u[1] = qa.y; aq.u[2] = qa.z; aq.u[3] = qa.w;
        aq.u[4] = qc.x; aq.u[5] = qc.y; aq.u[6] = qc.z; aq.u[7] = qc.w;
    }

    v8f o0 = {}, o1 = {};
    float m[8], l[8];
    #pragma unroll
    for (int r = 0; r < 8; r++) { m[r] = -1e30f; l[r] = 0.f; }

    const int sj = tid >> 3;         // staging: key 0..31
    const int sd = (tid & 7) * 4;    // staging: hd 0,4,...,28
    const unsigned short* kbase = kbuf + (size_t)bh * NPIX * HD + (size_t)sj * HD + sd;
    const unsigned short* vbase = vbuf + (size_t)bh * NPIX * HD + (size_t)sj * HD + sd;

    auto stageKV = [&](int buf, uint2 kr, uint2 vr) {
        unsigned* lk32 = (unsigned*)lds_k[buf];
        lk32[sj * 20 + (sd >> 1) + 0] = kr.x;
        lk32[sj * 20 + (sd >> 1) + 1] = kr.y;
        unsigned short* lv16 = lds_v[buf];
        lv16[(sd + 0) * 40 + sj] = (unsigned short)(vr.x & 0xffff);
        lv16[(sd + 1) * 40 + sj] = (unsigned short)(vr.x >> 16);
        lv16[(sd + 2) * 40 + sj] = (unsigned short)(vr.y & 0xffff);
        lv16[(sd + 3) * 40 + sj] = (unsigned short)(vr.y >> 16);
    };

    unsigned short* pw = pbuf + wid * 16 * 40;
    unsigned* pw32 = (unsigned*)pw;
    v8f zero = {};

    uint2 kreg = *(const uint2*)(kbase);
    uint2 vreg = *(const uint2*)(vbase);
    stageKV(0, kreg, vreg);
    __syncthreads();

    for (int it = 0; it < NPIX / 32; ++it) {
        const int cur = it & 1;
        if (it + 1 < NPIX / 32) {              // prefetch next 32-key chunk
            const size_t off = (size_t)(it + 1) * 32 * HD;
            kreg = *(const uint2*)(kbase + off);
            vreg = *(const uint2*)(vbase + off);
        }

        // S = Q . K^T  (B frag: col=key, k=hd dim)
        const unsigned* lk32 = (const unsigned*)lds_k[cur];
        const unsigned* lv32 = (const unsigned*)lds_v[cur];
        BF16Frag bk0, bk1;
        {
            const int kb2 = hi ? 4 : 0;
            const unsigned* p0 = lk32 + l16 * 20 + kb2;
            const unsigned* p1 = lk32 + (l16 + 16) * 20 + kb2;
            #pragma unroll
            for (int i = 0; i < 8; i++) { bk0.u[i] = p0[i + (i & 4)]; bk1.u[i] = p1[i + (i & 4)]; }
        }
        v8f s0 = __builtin_amdgcn_wmma_f32_16x16x32_bf16(false, aq.v, false, bk0.v,
                                                         (short)0, zero, false, false);
        v8f s1 = __builtin_amdgcn_wmma_f32_16x16x32_bf16(false, aq.v, false, bk1.v,
                                                         (short)0, zero, false, false);

        // online softmax (row r lives in a 16-lane group; xor-shuffle reductions)
        float p0v[8], p1v[8];
        #pragma unroll
        for (int r = 0; r < 8; r++) {
            float a = s0[r] * scale, c = s1[r] * scale;
            float mx = fmaxf(a, c);
            mx = fmaxf(mx, __shfl_xor(mx, 1));
            mx = fmaxf(mx, __shfl_xor(mx, 2));
            mx = fmaxf(mx, __shfl_xor(mx, 4));
            mx = fmaxf(mx, __shfl_xor(mx, 8));
            float mn  = fmaxf(m[r], mx);
            float fct = exp2f((m[r] - mn) * LOG2E);
            float pa  = exp2f((a - mn) * LOG2E);
            float pc  = exp2f((c - mn) * LOG2E);
            float rs  = pa + pc;
            rs += __shfl_xor(rs, 1); rs += __shfl_xor(rs, 2);
            rs += __shfl_xor(rs, 4); rs += __shfl_xor(rs, 8);
            l[r] = l[r] * fct + rs;
            m[r] = mn;
            o0[r] *= fct; o1[r] *= fct;
            p0v[r] = pa; p1v[r] = pc;
        }

        // C-layout P -> LDS -> A-layout P (wave-private; explicit dscnt wait)
        {
            const int prow = hi ? 8 : 0;
            __bf16* pwb = (__bf16*)pw;
            #pragma unroll
            for (int r = 0; r < 8; r++) {
                pwb[(prow + r) * 40 + l16]      = (__bf16)p0v[r];
                pwb[(prow + r) * 40 + l16 + 16] = (__bf16)p1v[r];
            }
        }
        asm volatile("s_wait_dscnt 0x0" ::: "memory");
        BF16Frag ap;
        {
            const unsigned* p = pw32 + l16 * 20;
            const int o4 = hi ? 4 : 0;
            #pragma unroll
            for (int i = 0; i < 4; i++) { ap.u[i] = p[o4 + i]; ap.u[4 + i] = p[8 + o4 + i]; }
        }

        // O += P . V   (B frag: col=hd, k=key; lds_v is [hd][key])
        BF16Frag bv0, bv1;
        {
            const int kb2 = hi ? 4 : 0;
            const unsigned* p0 = lv32 + l16 * 20 + kb2;
            const unsigned* p1 = lv32 + (l16 + 16) * 20 + kb2;
            #pragma unroll
            for (int i = 0; i < 8; i++) { bv0.u[i] = p0[i + (i & 4)]; bv1.u[i] = p1[i + (i & 4)]; }
        }
        o0 = __builtin_amdgcn_wmma_f32_16x16x32_bf16(false, ap.v, false, bv0.v,
                                                     (short)0, o0, false, false);
        o1 = __builtin_amdgcn_wmma_f32_16x16x32_bf16(false, ap.v, false, bv1.v,
                                                     (short)0, o1, false, false);

        if (it + 1 < NPIX / 32) stageKV(cur ^ 1, kreg, vreg);
        __syncthreads();
    }

    // normalize, write attn_out[b][h*32+hd][query] (bf16, proj-GEMM-friendly)
    U4p pk0, pk1;
    #pragma unroll
    for (int r = 0; r < 8; r += 2) {
        float i0 = 1.f / l[r], i1 = 1.f / l[r + 1];
        pk0.u[r >> 1] = pack2(o0[r] * i0, o0[r + 1] * i1);
        pk1.u[r >> 1] = pack2(o1[r] * i0, o1[r + 1] * i1);
    }
    const int qrun = q0 + (hi ? 8 : 0);
    unsigned short* d0 = attn_out + ((size_t)(b * DIM + h * HD + l16)      * NPIX + qrun);
    unsigned short* d1 = attn_out + ((size_t)(b * DIM + h * HD + l16 + 16) * NPIX + qrun);
    *(uint4*)d0 = pk0.q;
    *(uint4*)d1 = pk1.q;
}

// ---------------------------------------------------------------------------
// Kernel 3: output projection GEMM (fp32 out + bias), same tiling as kernel 1.
// ---------------------------------------------------------------------------
__global__ __launch_bounds__(256) void proj_gemm(
    const unsigned short* __restrict__ attn, const float* __restrict__ w,
    const float* __restrict__ bias, float* __restrict__ out)
{
    __shared__ unsigned short lds_a[2][64 * 40];
    __shared__ unsigned short lds_b[2][32 * 40];

    const int tid = threadIdx.x;
    const int wid = tid >> 5, lane = tid & 31;
    const int l16 = lane & 15;
    const bool hi = lane >= 16;
    const int b = blockIdx.z;
    const int nbase = blockIdx.x * 32;
    const int m0 = blockIdx.y * 64;
    const int mw = (wid >> 1) * 16;
    const int cw = (wid & 1) * 16;

    const int arow = tid >> 2;
    const int akg  = (tid & 3) * 8;
    const int bn   = tid & 31;
    const int bdg  = (tid >> 5) * 4;

    const float* asrc = w + (size_t)(m0 + arow) * DIM + akg;
    const unsigned short* bsrc = attn + ((size_t)b * DIM + bdg) * NPIX + nbase + bn;

    auto stageA = [&](int buf, float4 f0, float4 f1) {
        unsigned* dst = (unsigned*)lds_a[buf] + arow * 20 + (akg >> 1);
        dst[0] = pack2(f0.x, f0.y); dst[1] = pack2(f0.z, f0.w);
        dst[2] = pack2(f1.x, f1.y); dst[3] = pack2(f1.z, f1.w);
    };
    auto stageB = [&](int buf, unsigned t0, unsigned t1, unsigned t2, unsigned t3) {
        unsigned* dst = (unsigned*)lds_b[buf] + bn * 20 + (bdg >> 1);
        dst[0] = t0 | (t1 << 16); dst[1] = t2 | (t3 << 16);
    };

    v8f acc = {};
    float4 fa0 = *(const float4*)(asrc);
    float4 fa1 = *(const float4*)(asrc + 4);
    unsigned t0 = bsrc[0 * NPIX], t1 = bsrc[1 * NPIX], t2 = bsrc[2 * NPIX], t3 = bsrc[3 * NPIX];
    stageA(0, fa0, fa1); stageB(0, t0, t1, t2, t3);
    __syncthreads();

    #pragma unroll 2
    for (int kci = 0; kci < 8; ++kci) {
        const int cur = kci & 1;
        if (kci < 7) {
            const int kc = (kci + 1) * 32;
            fa0 = *(const float4*)(asrc + kc);
            fa1 = *(const float4*)(asrc + kc + 4);
            t0 = bsrc[(size_t)(kc + 0) * NPIX]; t1 = bsrc[(size_t)(kc + 1) * NPIX];
            t2 = bsrc[(size_t)(kc + 2) * NPIX]; t3 = bsrc[(size_t)(kc + 3) * NPIX];
        }
        BF16Frag af, bfr;
        {
            const int o4 = hi ? 4 : 0;
            const unsigned* p = (const unsigned*)lds_a[cur] + (mw + l16) * 20;
            #pragma unroll
            for (int i = 0; i < 4; i++) { af.u[i] = p[o4 + i]; af.u[4 + i] = p[8 + o4 + i]; }
        }
        {
            const unsigned* p = (const unsigned*)lds_b[cur] + (cw + l16) * 20 + (hi ? 4 : 0);
            #pragma unroll
            for (int i = 0; i < 8; i++) bfr.u[i] = p[i + (i & 4)];
        }
        acc = __builtin_amdgcn_wmma_f32_16x16x32_bf16(false, af.v, false, bfr.v,
                                                      (short)0, acc, false, false);
        if (kci < 7) { stageA(cur ^ 1, fa0, fa1); stageB(cur ^ 1, t0, t1, t2, t3); }
        __syncthreads();
    }

    const int obase = m0 + mw + (hi ? 8 : 0);
    const int n = nbase + cw + l16;
    #pragma unroll
    for (int r = 0; r < 8; r++)
        out[(size_t)(b * DIM + obase + r) * NPIX + n] = acc[r] + bias[obase + r];
}

// ---------------------------------------------------------------------------
extern "C" void kernel_launch(void* const* d_in, const int* in_sizes, int n_in,
                              void* d_out, int out_size, void* d_ws, size_t ws_size,
                              hipStream_t stream) {
    (void)in_sizes; (void)n_in; (void)out_size; (void)ws_size;
    const float* x      = (const float*)d_in[0];
    const float* w_qkv  = (const float*)d_in[1];
    const float* b_qkv  = (const float*)d_in[2];
    const float* w_proj = (const float*)d_in[3];
    const float* b_proj = (const float*)d_in[4];
    float* out = (float*)d_out;

    unsigned short* ws = (unsigned short*)d_ws;
    const size_t per = (size_t)NB * NH * NPIX * HD;   // 2M bf16 elems each
    unsigned short* qb   = ws;
    unsigned short* kb   = ws + per;
    unsigned short* vb   = ws + 2 * per;
    unsigned short* attn = ws + 3 * per;              // total 16 MB of ws

    dim3 blk(256);
    qkv_gemm  <<<dim3(NPIX / 32, 768 / 64, NB), blk, 0, stream>>>(x, w_qkv, b_qkv, qb, kb, vb);
    attn_kernel<<<dim3(NPIX / 128, NH, NB),     blk, 0, stream>>>(qb, kb, vb, attn);
    proj_gemm <<<dim3(NPIX / 32, DIM / 64, NB), blk, 0, stream>>>(attn, w_proj, b_proj, out);
}
